// kmapGNN_31671088840981
// MI455X (gfx1250) — matrix operand order repaired and checked by reading, compiled, not run on hardware
//
#include <hip/hip_runtime.h>
#include <hip/hip_bf16.h>
#include <math.h>

typedef __attribute__((ext_vector_type(16))) _Float16 v16h;
typedef __attribute__((ext_vector_type(8)))  float    v8f;

#define LEAK 0.02f

// Generic batched GEMM on WMMA f16->f32:
//   C[z] = act( alpha * A[z](MxK) * op(B[z]) + bias )
// A row-major [M,K] ld lda.
// BNK==1: B is [N,K] ld ldb  (C = A * B^T, torch weight layout)
// BNK==0: B is [K,N] ld ldb  (C = A * B, e.g. P*V)
// Batch z: zo=z/innerCnt, zi=z%innerCnt; base += zo*s?o + zi*s?i (uniform).
// One wave32 computes a 16x32 C tile (two 16x16 WMMA accumulators).
// Requires M % 16 == 0 (true at all call sites). N/K edges handled by
// index clamping + select-to-zero: no EXEC divergence in the K loop.
// Bases stay uniform (SGPR) and lanes index with 32-bit element offsets so
// loads lower to saddr+voffset form with immediate offsets.
template <int BNK>
__global__ __launch_bounds__(32)
void gemm_wmma_f16(const float* __restrict__ A, int lda, long sAo, long sAi,
                   const float* __restrict__ B, int ldb, long sBo, long sBi,
                   const float* __restrict__ bias,
                   float* __restrict__ C, int ldc, long sCo, long sCi,
                   int M, int N, int K, int innerCnt,
                   float alpha, int act)
{
    const int lane = threadIdx.x;
    const int z    = blockIdx.z;
    const int zo   = z / innerCnt;
    const int zi   = z % innerCnt;
    const float* Ab = A + zo * sAo + zi * sAi;   // uniform bases (SGPR)
    const float* Bb = B + zo * sBo + zi * sBi;
    float*       Cb = C + zo * sCo + zi * sCi;

    const int mt   = blockIdx.y * 16;
    const int nt   = blockIdx.x * 32;
    const int l15  = lane & 15;
    const int hgrp = lane >> 4;            // 0: lanes 0-15, 1: lanes 16-31

    // A: lane supplies row M = mt+l15 (M%16==0 -> always in range)
    //    a[j] <-> K = kb + hgrp*16 + j
    const int aBase = (mt + l15) * lda + hgrp * 16;

    // B: element(n,k) at index  (BNK ? n*ldb + k : k*ldb + n)
    //    b[j] <-> K = kb + hgrp*16 + j
    const int colB0 = min(nt + l15,      N - 1);
    const int colB1 = min(nt + 16 + l15, N - 1);
    const int bStep = BNK ? 1 : ldb;                    // index step per +1 in K
    const int bBase0 = BNK ? colB0 * ldb + hgrp * 16
                           : colB0 + hgrp * 16 * ldb;
    const int bBase1 = BNK ? colB1 * ldb + hgrp * 16
                           : colB1 + hgrp * 16 * ldb;

    v8f acc0 = {};
    v8f acc1 = {};

    const int kFull = K & ~31;
    int kb = 0;
    for (; kb < kFull; kb += 32) {
        v16h a;
        const int ao = aBase + kb;
        #pragma unroll
        for (int j = 0; j < 8; ++j) {
            a[j]     = (_Float16)Ab[ao + j];
            a[8 + j] = (_Float16)Ab[ao + 8 + j];
        }
        v16h b0, b1;
        const int bo0 = bBase0 + kb * bStep;
        const int bo1 = bBase1 + kb * bStep;
        #pragma unroll
        for (int j = 0; j < 16; ++j) {
            b0[j] = (_Float16)Bb[bo0 + j * bStep];
            b1[j] = (_Float16)Bb[bo1 + j * bStep];
        }
        acc0 = __builtin_amdgcn_wmma_f32_16x16x32_f16(
                   false, a, false, b0, (short)0, acc0, false, false);
        acc1 = __builtin_amdgcn_wmma_f32_16x16x32_f16(
                   false, a, false, b1, (short)0, acc1, false, false);
    }
    if (kb < K) {
        // K tail: clamped unconditional loads + select-to-zero (no EXEC change)
        v16h a;
        #pragma unroll
        for (int j = 0; j < 8; ++j) {
            const int k0 = kb + hgrp * 16 + j;
            const int k1 = k0 + 8;
            const float v0 = Ab[aBase - hgrp * 16 + min(k0, K - 1)];
            const float v1 = Ab[aBase - hgrp * 16 + min(k1, K - 1)];
            a[j]     = (k0 < K) ? (_Float16)v0 : (_Float16)0.0f;
            a[8 + j] = (k1 < K) ? (_Float16)v1 : (_Float16)0.0f;
        }
        v16h b0, b1;
        const int bRow0 = BNK ? colB0 * ldb : colB0;
        const int bRow1 = BNK ? colB1 * ldb : colB1;
        #pragma unroll
        for (int j = 0; j < 16; ++j) {
            const int k  = kb + hgrp * 16 + j;
            const int kc = min(k, K - 1);
            const float v0 = Bb[bRow0 + kc * bStep];
            const float v1 = Bb[bRow1 + kc * bStep];
            b0[j] = (k < K) ? (_Float16)v0 : (_Float16)0.0f;
            b1[j] = (k < K) ? (_Float16)v1 : (_Float16)0.0f;
        }
        acc0 = __builtin_amdgcn_wmma_f32_16x16x32_f16(
                   false, a, false, b0, (short)0, acc0, false, false);
        acc1 = __builtin_amdgcn_wmma_f32_16x16x32_f16(
                   false, a, false, b1, (short)0, acc1, false, false);
    }

    // C tile: acc[i] -> row = mt + hgrp*8 + i
    const int col0 = nt + l15;
    const int col1 = nt + 16 + l15;
    const float bv0 = (bias && col0 < N) ? bias[col0] : 0.0f;
    const float bv1 = (bias && col1 < N) ? bias[col1] : 0.0f;
    const int cBase = (mt + hgrp * 8) * ldc;
    #pragma unroll
    for (int i = 0; i < 8; ++i) {
        const int ro = cBase + i * ldc;
        if (col0 < N) {
            float v = acc0[i] * alpha + bv0;
            if (act) v = (v > 0.0f) ? v : v * LEAK;
            Cb[ro + col0] = v;
        }
        if (col1 < N) {
            float v = acc1[i] * alpha + bv1;
            if (act) v = (v > 0.0f) ? v : v * LEAK;
            Cb[ro + col1] = v;
        }
    }
}

// Row softmax, one wave32 per row of length L.
__global__ __launch_bounds__(32)
void softmax_rows(float* __restrict__ X, int L)
{
    float* row = X + (long)blockIdx.x * L;
    const int lane = threadIdx.x;
    float m = -3.4e38f;
    for (int i = lane; i < L; i += 32) m = fmaxf(m, row[i]);
    #pragma unroll
    for (int o = 16; o > 0; o >>= 1) m = fmaxf(m, __shfl_xor(m, o, 32));
    float s = 0.0f;
    for (int i = lane; i < L; i += 32) s += __expf(row[i] - m);
    #pragma unroll
    for (int o = 16; o > 0; o >>= 1) s += __shfl_xor(s, o, 32);
    const float inv = 1.0f / s;
    for (int i = lane; i < L; i += 32) row[i] = __expf(row[i] - m) * inv;
}

// Y[s,e] = mean_l X[s,l,e]
__global__ void mean_rows(const float* __restrict__ X, float* __restrict__ Y,
                          int S, int L, int E)
{
    const long idx = (long)blockIdx.x * blockDim.x + threadIdx.x;
    if (idx >= (long)S * E) return;
    const int s = (int)(idx / E);
    const int e = (int)(idx % E);
    const float* p = X + (long)s * L * E + e;
    float acc = 0.0f;
    for (int l = 0; l < L; ++l) acc += p[(long)l * E];
    Y[idx] = acc * (1.0f / (float)L);
}

// Final 128->2 linear + softmax, one thread per batch row.
__global__ __launch_bounds__(32)
void final_head(const float* __restrict__ Z, const float* __restrict__ W,
                const float* __restrict__ bias, float* __restrict__ out,
                int Bn, int K)
{
    const int i = blockIdx.x * blockDim.x + threadIdx.x;
    if (i >= Bn) return;
    const float* z = Z + (long)i * K;
    float a0 = bias[0], a1 = bias[1];
    for (int k = 0; k < K; ++k) {
        a0 += z[k] * W[k];
        a1 += z[k] * W[K + k];
    }
    const float m  = fmaxf(a0, a1);
    const float e0 = __expf(a0 - m), e1 = __expf(a1 - m);
    const float inv = 1.0f / (e0 + e1);
    out[2 * i]     = e0 * inv;
    out[2 * i + 1] = e1 * inv;
}

extern "C" void kernel_launch(void* const* d_in, const int* in_sizes, int n_in,
                              void* d_out, int out_size, void* d_ws, size_t ws_size,
                              hipStream_t stream)
{
    (void)in_sizes; (void)n_in; (void)out_size; (void)ws_size;

    const float* x2      = (const float*)d_in[0];   // [32,16,128,27]
    const float* W1      = (const float*)d_in[1];   // [300,27]
    const float* b1      = (const float*)d_in[2];
    const float* A1_in_w = (const float*)d_in[3];   // [900,300]
    const float* A1_in_b = (const float*)d_in[4];
    const float* A1_o_w  = (const float*)d_in[5];   // [300,300]
    const float* A1_o_b  = (const float*)d_in[6];
    const float* W2      = (const float*)d_in[7];   // [300,300]
    const float* b2      = (const float*)d_in[8];
    const float* A2_in_w = (const float*)d_in[9];   // [900,300]
    const float* A2_in_b = (const float*)d_in[10];
    const float* A2_o_w  = (const float*)d_in[11];  // [300,300]
    const float* A2_o_b  = (const float*)d_in[12];
    const float* C1_w    = (const float*)d_in[13];  // [128,300]
    const float* C1_b    = (const float*)d_in[14];
    const float* C2_w    = (const float*)d_in[15];  // [128,128]
    const float* C2_b    = (const float*)d_in[16];
    const float* C3_w    = (const float*)d_in[17];  // [2,128]
    const float* C3_b    = (const float*)d_in[18];
    float* out = (float*)d_out;                     // [32,2]

    // Sizes
    const int Bn = 32, N1 = 16, N2 = 128, F = 27;
    const int E = 300, H = 6, D = 50;               // E/H = 50
    const int S1 = Bn * N1;                         // 512 sequences, L=128
    const int L1 = N2;
    const int M1 = S1 * L1;                         // 65536 rows
    const int S2 = Bn, L2 = N1;
    const int M2 = S2 * L2;                         // 512 rows
    const float isqd = 0.14142135623730951f;        // 1/sqrt(50)

    // Workspace partition (floats)
    float* P0 = (float*)d_ws;                       // H1, later attnout  [65536,300]
    float* P1 = P0 + (size_t)M1 * E;                // QKV [65536,900]; layer-2 arena
    float* P2 = P1 + (size_t)M1 * 3 * E;            // scores [3072,128,128]; later proj1
    float* H1      = P0;
    float* QKV     = P1;
    float* scores  = P2;
    float* attnout = P0;                            // reuses H1 (dead after QKV)
    float* proj1   = P2;                            // reuses scores (dead after attnout)
    // layer-2 arena inside P1 (QKV dead after attnout)
    float* meanh   = P1;                            // [512,300]
    float* h2      = P1 + 153600;                   // [512,300]
    float* qkv2    = P1 + 307200;                   // [512,900]
    float* scores2 = P1 + 768000;                   // [192,16,16]
    float* att2    = P1 + 817152;                   // [512,300]
    float* proj2   = P1 + 970752;                   // [512,300]
    float* emb     = P1 + 1124352;                  // [32,300]
    float* z1      = P1 + 1133952;                  // [32,128]
    float* z2      = P1 + 1138048;                  // [32,128]

    const dim3 wave(32);

    // ---- GraphLayer 1 (S=512, L=128) ----
    // H1 = lrelu(x2 @ W1^T + b1)   [65536,300]
    gemm_wmma_f16<1><<<dim3(10, 4096, 1), wave, 0, stream>>>(
        x2, F, 0, 0,  W1, F, 0, 0,  b1,  H1, E, 0, 0,
        M1, E, F, 1, 1.0f, 1);
    // QKV = H1 @ A1_in_w^T + A1_in_b   [65536,900]
    gemm_wmma_f16<1><<<dim3(29, 4096, 1), wave, 0, stream>>>(
        H1, E, 0, 0,  A1_in_w, E, 0, 0,  A1_in_b,  QKV, 3 * E, 0, 0,
        M1, 3 * E, E, 1, 1.0f, 0);
    // scores[s,h] = (Q Kt)/sqrt(d)   batch 3072, M=N=128, K=50
    gemm_wmma_f16<1><<<dim3(4, 8, S1 * H), wave, 0, stream>>>(
        QKV,       3 * E, (long)L1 * 3 * E, D,
        QKV + E,   3 * E, (long)L1 * 3 * E, D,
        nullptr,
        scores, L1, (long)H * L1 * L1, (long)L1 * L1,
        L1, L1, D, H, isqd, 0);
    softmax_rows<<<dim3(S1 * H * L1), wave, 0, stream>>>(scores, L1);
    // attnout[s,:,h*50+j] = P V     batch 3072, M=128, N=50, K=128 (B is [K,N])
    gemm_wmma_f16<0><<<dim3(2, 8, S1 * H), wave, 0, stream>>>(
        scores,      L1, (long)H * L1 * L1, (long)L1 * L1,
        QKV + 2 * E, 3 * E, (long)L1 * 3 * E, D,
        nullptr,
        attnout, E, (long)L1 * E, D,
        L1, D, L1, H, 1.0f, 0);
    // proj1 = attnout @ A1_out_w^T + b   [65536,300]
    gemm_wmma_f16<1><<<dim3(10, 4096, 1), wave, 0, stream>>>(
        attnout, E, 0, 0,  A1_o_w, E, 0, 0,  A1_o_b,  proj1, E, 0, 0,
        M1, E, E, 1, 1.0f, 0);
    // mean over L=128 -> meanh [512,300]
    mean_rows<<<dim3((S1 * E + 255) / 256), dim3(256), 0, stream>>>(
        proj1, meanh, S1, L1, E);

    // ---- GraphLayer 2 (S=32, L=16) ----
    gemm_wmma_f16<1><<<dim3(10, 32, 1), wave, 0, stream>>>(
        meanh, E, 0, 0,  W2, E, 0, 0,  b2,  h2, E, 0, 0,
        M2, E, E, 1, 1.0f, 1);
    gemm_wmma_f16<1><<<dim3(29, 32, 1), wave, 0, stream>>>(
        h2, E, 0, 0,  A2_in_w, E, 0, 0,  A2_in_b,  qkv2, 3 * E, 0, 0,
        M2, 3 * E, E, 1, 1.0f, 0);
    gemm_wmma_f16<1><<<dim3(1, 1, S2 * H), wave, 0, stream>>>(
        qkv2,     3 * E, (long)L2 * 3 * E, D,
        qkv2 + E, 3 * E, (long)L2 * 3 * E, D,
        nullptr,
        scores2, L2, (long)H * L2 * L2, (long)L2 * L2,
        L2, L2, D, H, isqd, 0);
    softmax_rows<<<dim3(S2 * H * L2), wave, 0, stream>>>(scores2, L2);
    gemm_wmma_f16<0><<<dim3(2, 1, S2 * H), wave, 0, stream>>>(
        scores2,      L2, (long)H * L2 * L2, (long)L2 * L2,
        qkv2 + 2 * E, 3 * E, (long)L2 * 3 * E, D,
        nullptr,
        att2, E, (long)L2 * E, D,
        L2, D, L2, H, 1.0f, 0);
    gemm_wmma_f16<1><<<dim3(10, 32, 1), wave, 0, stream>>>(
        att2, E, 0, 0,  A2_o_w, E, 0, 0,  A2_o_b,  proj2, E, 0, 0,
        M2, E, E, 1, 1.0f, 0);
    mean_rows<<<dim3((S2 * E + 255) / 256), dim3(256), 0, stream>>>(
        proj2, emb, S2, L2, E);

    // ---- Classifier MLP ----
    gemm_wmma_f16<1><<<dim3(4, 2, 1), wave, 0, stream>>>(
        emb, E, 0, 0,  C1_w, E, 0, 0,  C1_b,  z1, 128, 0, 0,
        Bn, 128, E, 1, 1.0f, 1);
    gemm_wmma_f16<1><<<dim3(4, 2, 1), wave, 0, stream>>>(
        z1, 128, 0, 0,  C2_w, 128, 0, 0,  C2_b,  z2, 128, 0, 0,
        Bn, 128, 128, 1, 1.0f, 1);
    final_head<<<dim3(1), wave, 0, stream>>>(z2, C3_w, C3_b, out, Bn, 128);
}
